// LoRA_SNN_74423193305329
// MI455X (gfx1250) — matrix operand order repaired and checked by reading, not compiled, not run on hardware
//
#include <hip/hip_runtime.h>
#include <hip/hip_bf16.h>
#include <stdint.h>

// ---------------------------------------------------------------------------
// LoRA-SNN on MI455X (gfx1250, wave32, WMMA).
//
//   W1eff = W1^T + A1@B1   (4096x4096)  -- LoRA folded once
//   W2eff = W2^T + A2@B2   (4096x1024)
//   cur1  = x @ W1eff + b1              -- one f16 WMMA GEMM, f32 accum
//   loop t=0..24:
//     lif1: m1 = .9*m1 + cur1 - (m1>1); spk1 = (m1>1)   -> packed f16 A-frags
//     gemm2+lif2 fused: cur2 = spk1 @ W2eff + b2; m2 update; write spk2/mem2
//
// Fragment-major operand packing matches the CDNA5 16x16x32 f16 WMMA VGPR
// layouts (ISA 7.12.2): every WMMA operand is two aligned global_load_b128.
// All accesses forced to address_space(1) => global_load/store (no flat,
// no DScnt coupling). Recorded outputs use nontemporal stores so the
// ~88 MB hot set (W2eff+spk1+cur1+m1) stays resident in the 192 MB L2.
// ---------------------------------------------------------------------------

typedef __attribute__((ext_vector_type(16))) _Float16 v16h;
typedef __attribute__((ext_vector_type(8)))  float    v8f;

#define AS1 __attribute__((address_space(1)))

template <typename T>
__device__ __forceinline__ T ldg(const void* p) {
    return *(const AS1 T*)(uintptr_t)p;
}
template <typename T>
__device__ __forceinline__ void stg(void* p, T v) {
    *(AS1 T*)(uintptr_t)p = v;
}
template <typename T>
__device__ __forceinline__ void stg_nt(void* p, T v) {
    __builtin_nontemporal_store(v, (AS1 T*)(uintptr_t)p);
}

#define B_SZ    2048
#define D_IN    4096
#define D_H     4096
#define D_OUT   1024
#define T_STEPS 25
#define BETA    0.9f
#define THR     1.0f

// ---- packed-layout index helpers (element offsets into f16 arrays) --------
// A-fragment (16x32, MxK), ISA 7.12.2.
__device__ __forceinline__ size_t a_pack_idx(int m, int k, int K) {
    int kp   = k & 31;
    int lane = (m & 15) + ((kp & 8) ? 16 : 0);
    int e    = (kp & 7) + ((kp & 16) ? 8 : 0);
    size_t tile = (size_t)(m >> 4) * (size_t)(K >> 5) + (size_t)(k >> 5);
    return tile * 512 + (size_t)lane * 16 + (size_t)e;
}
// B-fragment (32x16, KxN): same k order as A-fragment (lanes 0-15 K=0..7,16..23; lanes 16-31 K=8..15,24..31).
__device__ __forceinline__ size_t b_pack_idx(int k, int n, int N) {
    int lane = (n & 15) + ((k & 8) ? 16 : 0);
    int e    = (k & 7) + ((k & 16) ? 8 : 0);
    size_t tile = (size_t)(k >> 5) * (size_t)(N >> 4) + (size_t)(n >> 4);
    return tile * 512 + (size_t)lane * 16 + (size_t)e;
}

// ---- prep: W_eff = W^T + A@B, packed f16 B-fragment layout ----------------
// W is [N x K] row-major (torch-style weight), Alo [K x 16], Blo [16 x N].
__global__ __launch_bounds__(256)
void prep_weff_kernel(const float* __restrict__ W, const float* __restrict__ Alo,
                      const float* __restrict__ Blo, _Float16* __restrict__ Bpk,
                      int N) {
    size_t i = (size_t)blockIdx.x * blockDim.x + threadIdx.x;   // over K*N
    int k = (int)(i & 4095);          // K == 4096 for both layers
    int n = (int)(i >> 12);
    if (n >= N) return;
    float v = ldg<float>(W + (size_t)n * 4096 + k);
    float s = 0.f;
#pragma unroll
    for (int r = 0; r < 16; ++r)
        s += ldg<float>(Alo + (size_t)k * 16 + r) * ldg<float>(Blo + (size_t)r * N + n);
    v += s;                           // ALPHA == 1.0
    stg<_Float16>(Bpk + b_pack_idx(k, n, N), (_Float16)v);
}

// ---- prep: pack x (f32 row-major [M x K]) into f16 A-fragment layout ------
__global__ __launch_bounds__(256)
void pack_x_kernel(const float* __restrict__ x, _Float16* __restrict__ Apk) {
    size_t i = (size_t)blockIdx.x * blockDim.x + threadIdx.x;   // over M*K
    int m = (int)(i >> 12);           // K == 4096
    int k = (int)(i & 4095);
    stg<_Float16>(Apk + a_pack_idx(m, k, D_IN), (_Float16)ldg<float>(x + i));
}

// ---- per-timestep LIF on layer 1; emits spk1 in packed A layout -----------
__global__ __launch_bounds__(256)
void lif1_kernel(const float* __restrict__ cur1, float* __restrict__ m1,
                 _Float16* __restrict__ spk1pk, int t) {
    size_t i = (size_t)blockIdx.x * blockDim.x + threadIdx.x;   // over B*D_H
    int m = (int)(i >> 12);
    int k = (int)(i & 4095);
    float cur   = ldg<float>(cur1 + i);
    float mold  = (t == 0) ? 0.f : ldg<float>(m1 + i);
    float reset = (mold - THR > 0.f) ? 1.f : 0.f;
    float mn    = BETA * mold + cur - reset * THR;
    stg<float>(m1 + i, mn);
    float spk   = (mn - THR > 0.f) ? 1.f : 0.f;
    stg<_Float16>(spk1pk + a_pack_idx(m, k, D_H), (_Float16)spk);
}

// ---- WMMA GEMM: C[MxN] = Apk @ Bpk + bias, optional fused LIF epilogue ----
template<int BM, int BN, int WAVES_M, int WAVES_N, int WM, int WN, bool LIF>
__global__ __launch_bounds__(WAVES_M * WAVES_N * 32)
void gemm_wmma_kernel(const _Float16* __restrict__ Apk,
                      const _Float16* __restrict__ Bpk,
                      const float* __restrict__ bias,
                      float* __restrict__ Cout,
                      float* __restrict__ mem,
                      float* __restrict__ out_spk,
                      float* __restrict__ out_mem,
                      int M, int N, int K, int t) {
    const int lane = threadIdx.x & 31;
    const int wave = threadIdx.x >> 5;
    const int wm   = wave / WAVES_N;
    const int wn   = wave % WAVES_N;
    const int KT   = K >> 5;                       // number of 32-wide k-tiles
    const int NT   = N >> 4;                       // number of 16-wide n-tiles
    const int mt0  = blockIdx.y * (BM / 16) + wm * WM;
    const int nt0  = blockIdx.x * (BN / 16) + wn * WN;

    v8f acc[WM][WN] = {};

    const _Float16* aptr[WM];
    const _Float16* bptr[WN];
#pragma unroll
    for (int i = 0; i < WM; ++i)
        aptr[i] = Apk + (size_t)(mt0 + i) * KT * 512 + (size_t)lane * 16;
#pragma unroll
    for (int j = 0; j < WN; ++j)
        bptr[j] = Bpk + (size_t)(nt0 + j) * 512 + (size_t)lane * 16;
    const size_t bstride = (size_t)NT * 512;

    for (int kt = 0; kt < KT; ++kt) {
        v16h a[WM], b[WN];
#pragma unroll
        for (int i = 0; i < WM; ++i) { a[i] = ldg<v16h>(aptr[i]); aptr[i] += 512; }
#pragma unroll
        for (int j = 0; j < WN; ++j) { b[j] = ldg<v16h>(bptr[j]); bptr[j] += bstride; }
#pragma unroll
        for (int i = 0; i < WM; ++i)
#pragma unroll
            for (int j = 0; j < WN; ++j)
                acc[i][j] = __builtin_amdgcn_wmma_f32_16x16x32_f16(
                    false, a[i], false, b[j], (short)0, acc[i][j], false, false);
    }

    // epilogue: C/D layout (ISA 7.12.2): n = 16*nt + (lane&15),
    //           m = 16*mt + r + (lane>=16 ? 8 : 0)
    const int nl   = lane & 15;
    const int moff = (lane >> 4) << 3;
#pragma unroll
    for (int i = 0; i < WM; ++i) {
#pragma unroll
        for (int j = 0; j < WN; ++j) {
            const int n  = (nt0 + j) * 16 + nl;
            const float bv = ldg<float>(bias + n);
#pragma unroll
            for (int r = 0; r < 8; ++r) {
                const int mrow = (mt0 + i) * 16 + moff + r;
                const size_t idx = (size_t)mrow * N + n;
                const float val = acc[i][j][r] + bv;
                if (LIF) {
                    float mold  = (t == 0) ? 0.f : ldg<float>(mem + idx);
                    float reset = (mold - THR > 0.f) ? 1.f : 0.f;
                    float mn    = BETA * mold + val - reset * THR;
                    stg<float>(mem + idx, mn);
                    float spk   = (mn - THR > 0.f) ? 1.f : 0.f;
                    const size_t oidx = (size_t)t * M * N + idx;
                    stg_nt<float>(out_spk + oidx, spk);
                    stg_nt<float>(out_mem + oidx, mn);
                } else {
                    stg<float>(Cout + idx, val);
                }
            }
        }
    }
}

extern "C" void kernel_launch(void* const* d_in, const int* in_sizes, int n_in,
                              void* d_out, int out_size, void* d_ws, size_t ws_size,
                              hipStream_t stream) {
    const float* x  = (const float*)d_in[0];
    const float* W1 = (const float*)d_in[1];
    const float* b1 = (const float*)d_in[2];
    const float* W2 = (const float*)d_in[3];
    const float* b2 = (const float*)d_in[4];
    const float* A1 = (const float*)d_in[5];
    const float* B1 = (const float*)d_in[6];
    const float* A2 = (const float*)d_in[7];
    const float* B2 = (const float*)d_in[8];

    // workspace carve-up (~151 MB total)
    char* ws = (char*)d_ws;
    size_t off = 0;
    _Float16* W1pk   = (_Float16*)(ws + off); off += (size_t)D_IN * D_H  * 2; // 32 MB
    _Float16* W2pk   = (_Float16*)(ws + off); off += (size_t)D_H  * D_OUT* 2; //  8 MB
    _Float16* Xpk    = (_Float16*)(ws + off); off += (size_t)B_SZ * D_IN * 2; // 16 MB
    float*    cur1   = (float*)   (ws + off); off += (size_t)B_SZ * D_H  * 4; // 32 MB
    float*    m1     = (float*)   (ws + off); off += (size_t)B_SZ * D_H  * 4; // 32 MB
    _Float16* spk1pk = (_Float16*)(ws + off); off += (size_t)B_SZ * D_H  * 2; // 16 MB
    float*    m2     = (float*)   (ws + off); off += (size_t)B_SZ * D_OUT* 4; //  8 MB

    float* out_spk = (float*)d_out;
    float* out_mem = out_spk + (size_t)T_STEPS * B_SZ * D_OUT;

    // 1) fold LoRA into effective weights (packed f16 B-fragment layout)
    prep_weff_kernel<<<dim3((D_IN * D_H ) / 256), dim3(256), 0, stream>>>(W1, A1, B1, W1pk, D_H);
    prep_weff_kernel<<<dim3((D_IN * D_OUT)/ 256), dim3(256), 0, stream>>>(W2, A2, B2, W2pk, D_OUT);

    // 2) pack x into f16 A-fragment layout
    pack_x_kernel<<<dim3(((size_t)B_SZ * D_IN) / 256), dim3(256), 0, stream>>>(x, Xpk);

    // 3) cur1 = Xpk @ W1pk + b1   (2048x4096x4096, f16 WMMA, f32 accum)
    //    block 128x128, 8 waves (2x4), wave tile 64x32 (4x2 WMMA tiles)
    gemm_wmma_kernel<128, 128, 2, 4, 4, 2, false>
        <<<dim3(D_H / 128, B_SZ / 128), dim3(256), 0, stream>>>(
            Xpk, W1pk, b1, cur1, nullptr, nullptr, nullptr,
            B_SZ, D_H, D_IN, 0);

    // 4) time loop
    for (int t = 0; t < T_STEPS; ++t) {
        lif1_kernel<<<dim3(((size_t)B_SZ * D_H) / 256), dim3(256), 0, stream>>>(
            cur1, m1, spk1pk, t);
        // cur2 = spk1 @ W2pk + b2 fused with LIF-2 + output writes
        // block 128x64, 8 waves (4x2), wave tile 32x32 (2x2 WMMA tiles)
        gemm_wmma_kernel<128, 64, 4, 2, 2, 2, true>
            <<<dim3(D_OUT / 64, B_SZ / 128), dim3(256), 0, stream>>>(
                spk1pk, W2pk, b2, nullptr, m2, out_spk, out_mem,
                B_SZ, D_OUT, D_H, t);
    }
}
